// Attention_54142357733562
// MI455X (gfx1250) — matrix operand order repaired and checked by reading, not compiled, not run on hardware
//
#include <hip/hip_runtime.h>

// ---------------------------------------------------------------------------
// Linear (Taylor) attention for MI455X / gfx1250, bf16 WMMA + fp32 accum.
// Decomposition avoids materializing V:  kv = (Kn x^T) wv^T + bv*sum(Kn),
// value_sum = wv * sum(x) + N*bv.  Memory-bound target ~24us.
// Data movement: b128 vector loads for fp32 x, ASYNCcnt-tracked
// global<->LDS b128 async copies for all bf16 tile traffic.
// ---------------------------------------------------------------------------

typedef __attribute__((ext_vector_type(16))) __bf16 v16bf;
typedef __attribute__((ext_vector_type(8)))  float  v8f;

#define B_   4
#define C_   256
#define CQK  32
#define NPIX 65536
#define NB1  128   // pixels per block, pass 1
#define NB3  256   // pixels per block, pass 3

static __device__ __forceinline__ v8f wmma_bf16(v16bf a, v16bf b, v8f c) {
  return __builtin_amdgcn_wmma_f32_16x16x32_bf16(false, a, false, b,
                                                 (short)0, c, false, false);
}

// --- CDNA5 async global<->LDS (ASYNCcnt) ------------------------------------
static __device__ __forceinline__ unsigned lds_off(const void* p) {
  // AS(3) pointer value == byte offset within the workgroup LDS allocation,
  // which is what the async instructions' LDS-address VGPR expects.
  return (unsigned)(size_t)(__attribute__((address_space(3))) const void*)p;
}
static __device__ __forceinline__ void async_load_b128(void* lds_dst, const void* gsrc) {
  asm volatile("global_load_async_to_lds_b128 %0, %1, off"
               :: "v"(lds_off(lds_dst)),
                  "v"((unsigned long long)(size_t)gsrc)
               : "memory");
}
static __device__ __forceinline__ void async_store_b128(void* gdst, const void* lds_src) {
  asm volatile("global_store_async_from_lds_b128 %0, %1, off"
               :: "v"((unsigned long long)(size_t)gdst),
                  "v"(lds_off(lds_src))
               : "memory");
}
static __device__ __forceinline__ void wait_async0() {
  asm volatile("s_wait_asynccnt 0x0" ::: "memory");
}

// A (16x32 bf16, M x K), row-major source base[m*ld + k].
// lanes 0-15 -> M=lane, K=0..15; lanes 16-31 -> K=16..31.
static __device__ __forceinline__ v16bf load_A(const __bf16* base, int ld, int lane) {
  const int m  = lane & 15;
  const int kh = (lane >> 4) << 4;            // 0 or 16
  const __bf16* row = base + m * ld;
  v16bf a;
#pragma unroll
  for (int v = 0; v < 8; ++v) {
    const int kb = kh + 2 * v;
    a[2 * v]     = row[kb];
    a[2 * v + 1] = row[kb + 1];
  }
  return a;
}

// B (32x16 bf16, K x N), source element (k,n) = base[k*ld + n].
// lanes 0-15 -> N=lane, K=0..15; lanes 16-31 -> K=16..31.
static __device__ __forceinline__ v16bf load_B(const __bf16* base, int ld, int lane) {
  const int n  = lane & 15;
  const int kh = (lane >> 4) << 4;            // 0 or 16
  v16bf b;
#pragma unroll
  for (int v = 0; v < 8; ++v) {
    const int k = kh + 2 * v;
    b[2 * v]     = base[k * ld + n];
    b[2 * v + 1] = base[(k + 1) * ld + n];
  }
  return b;
}

// B (32x16 bf16) from transposed source: element (k,n) = base[n*ld + k].
static __device__ __forceinline__ v16bf load_Bt(const __bf16* base, int ld, int lane) {
  const int n  = lane & 15;
  const int kh = (lane >> 4) << 4;
  const __bf16* col = base + n * ld;
  v16bf b;
#pragma unroll
  for (int v = 0; v < 8; ++v) {
    const int k = kh + 2 * v;
    b[2 * v]     = col[k];
    b[2 * v + 1] = col[k + 1];
  }
  return b;
}

// ---------------------------------------------------------------------------
// Pass 1: per 128-pixel tile: Q/K projection (WMMA), l2-normalize, async-spill
// Qn, accumulate S = Kn x^T (WMMA), k_sum, x_sum via fp32 atomics.
// ---------------------------------------------------------------------------
__global__ __launch_bounds__(256)
void la_pass1(const float* __restrict__ x,
              const float* __restrict__ wq, const float* __restrict__ bq,
              const float* __restrict__ wk, const float* __restrict__ bk,
              __bf16* __restrict__ qn_g, float* __restrict__ S_g,
              float* __restrict__ ksum_g, float* __restrict__ xsum_g) {
  __shared__ __bf16 wq_s[CQK][C_];
  __shared__ __bf16 wk_s[CQK][C_];
  __shared__ __bf16 x_s[C_][NB1 + 8];
  __shared__ float  qf[CQK][NB1];
  __shared__ float  kf[CQK][NB1];
  __shared__ __bf16 qn_s[CQK][NB1 + 8];
  __shared__ __bf16 kn_s[CQK][NB1 + 8];
  __shared__ float  rnq[NB1], rnk[NB1];
  __shared__ float  bq_s[CQK], bk_s[CQK];

  const int tid  = threadIdx.x;
  const int wave = tid >> 5;
  const int lane = tid & 31;
  const int b    = blockIdx.y;
  const int p0   = blockIdx.x * NB1;

  if (tid < CQK) { bq_s[tid] = bq[tid]; bk_s[tid] = bk[tid]; }
  // weights: b128 loads (2048 float4 each)
  {
    const float4* wq4 = (const float4*)wq;
    const float4* wk4 = (const float4*)wk;
    for (int e = tid; e < CQK * C_ / 4; e += 256) {
      const int m = e >> 6;            // 64 float4 per 256-wide row
      const int q = (e & 63) * 4;
      const float4 a = wq4[e];
      const float4 c = wk4[e];
      wq_s[m][q + 0] = (__bf16)a.x; wq_s[m][q + 1] = (__bf16)a.y;
      wq_s[m][q + 2] = (__bf16)a.z; wq_s[m][q + 3] = (__bf16)a.w;
      wk_s[m][q + 0] = (__bf16)c.x; wk_s[m][q + 1] = (__bf16)c.y;
      wk_s[m][q + 2] = (__bf16)c.z; wk_s[m][q + 3] = (__bf16)c.w;
    }
  }
  // x tile: b128 loads, convert fp32 -> bf16 into LDS
  {
    const float4* xb4 = (const float4*)(x + (size_t)b * C_ * NPIX + p0);
    for (int e = tid; e < C_ * (NB1 / 4); e += 256) {
      const int c  = e >> 5;           // 32 float4 per 128-px row
      const int q  = e & 31;
      const float4 v = xb4[(size_t)c * (NPIX / 4) + q];
      const int px = q * 4;
      x_s[c][px + 0] = (__bf16)v.x; x_s[c][px + 1] = (__bf16)v.y;
      x_s[c][px + 2] = (__bf16)v.z; x_s[c][px + 3] = (__bf16)v.w;
    }
  }
  __syncthreads();

  // Q/K projection: [32 x 256] @ [256 x 128], 32 tiles over 8 waves.
  for (int t = wave; t < 32; t += 8) {
    const bool isQ = (t < 16);
    const int tt = t & 15;
    const int mt = tt >> 3;                 // 0..1 (M tile)
    const int nt = tt & 7;                  // 0..7 (pixel tile)
    const __bf16* Wb = isQ ? &wq_s[0][0] : &wk_s[0][0];
    v8f acc = {0.f, 0.f, 0.f, 0.f, 0.f, 0.f, 0.f, 0.f};
#pragma unroll
    for (int ks = 0; ks < 8; ++ks) {
      v16bf a  = load_A(Wb + (mt * 16) * C_ + ks * 32, C_, lane);
      v16bf bm = load_B(&x_s[ks * 32][nt * 16], NB1 + 8, lane);
      acc = wmma_bf16(a, bm, acc);
    }
    const int rowb = (lane >> 4) << 3;
    const int col  = lane & 15;
    float* OUT = isQ ? &qf[0][0] : &kf[0][0];
    const float* bias = isQ ? bq_s : bk_s;
#pragma unroll
    for (int r = 0; r < 8; ++r) {
      const int m = mt * 16 + rowb + r;
      OUT[m * NB1 + nt * 16 + col] = acc[r] + bias[m];
    }
  }
  __syncthreads();

  // per-pixel channel l2 norms
  if (tid < NB1) {
    float sq = 0.f, sk = 0.f;
#pragma unroll
    for (int m = 0; m < CQK; ++m) {
      const float q = qf[m][tid]; sq += q * q;
      const float k = kf[m][tid]; sk += k * k;
    }
    rnq[tid] = rsqrtf(sq);
    rnk[tid] = rsqrtf(sk);
  }
  __syncthreads();

  // normalized Q/K to LDS (bf16)
  for (int e = tid; e < CQK * NB1; e += 256) {
    const int m  = e >> 7;
    const int px = e & (NB1 - 1);
    qn_s[m][px] = (__bf16)(qf[m][px] * rnq[px]);
    kn_s[m][px] = (__bf16)(kf[m][px] * rnk[px]);
  }
  if (tid < CQK) {  // k_sum partial
    float s = 0.f;
    for (int px = 0; px < NB1; ++px) s += kf[tid][px] * rnk[px];
    atomicAdd(&ksum_g[b * CQK + tid], s);
  }
  {                 // x_sum partial (tid == channel)
    float s = 0.f;
    for (int px = 0; px < NB1; ++px) s += (float)x_s[tid][px];
    atomicAdd(&xsum_g[b * C_ + tid], s);
  }
  __syncthreads();

  // async-spill Qn tile LDS -> global (b128 chunks; completion via endpgm wait)
  for (int e = tid; e < CQK * (NB1 / 8); e += 256) {   // 512 chunks
    const int m  = e >> 4;
    const int pc = (e & 15) * 8;
    async_store_b128(qn_g + ((size_t)b * CQK + m) * NPIX + p0 + pc, &qn_s[m][pc]);
  }

  // S += Kn [32 x 128] @ x^T [128 x 256], 32 tiles over 8 waves, 4 k-steps.
  for (int t = wave; t < 32; t += 8) {
    const int mt = t >> 4;                  // 0..1
    const int ct = t & 15;                  // 0..15 (channel tile)
    v8f acc = {0.f, 0.f, 0.f, 0.f, 0.f, 0.f, 0.f, 0.f};
#pragma unroll
    for (int ks = 0; ks < 4; ++ks) {
      v16bf a  = load_A(&kn_s[mt * 16][ks * 32], NB1 + 8, lane);
      v16bf bm = load_Bt(&x_s[ct * 16][ks * 32], NB1 + 8, lane);
      acc = wmma_bf16(a, bm, acc);
    }
    const int rowb = (lane >> 4) << 3;
    const int col  = lane & 15;
#pragma unroll
    for (int r = 0; r < 8; ++r) {
      const int m = mt * 16 + rowb + r;
      const int c = ct * 16 + col;
      atomicAdd(&S_g[((size_t)b * CQK + m) * C_ + c], acc[r]);
    }
  }
}

// ---------------------------------------------------------------------------
// Pass 2 (tiny): kv^T[c][m] = sum_cc wv[c][cc] S[m][cc] + bv[c]*ksum[m];
// value_sum[c] = wv . xsum + N*bv[c]; ksum_eps = ksum + 1e-6.
// ---------------------------------------------------------------------------
__global__ __launch_bounds__(256)
void la_pass2(const float* __restrict__ wv, const float* __restrict__ bv,
              const float* __restrict__ S_g, const float* __restrict__ ksum_g,
              const float* __restrict__ xsum_g,
              __bf16* __restrict__ kvT_g, float* __restrict__ vsum_g,
              float* __restrict__ kse_g) {
  __shared__ float S_s[CQK * C_];
  __shared__ float xs_s[C_];
  __shared__ float ksr_s[CQK];
  const int tid = threadIdx.x;
  const int b   = blockIdx.x;

  for (int e = tid; e < CQK * C_; e += 256) S_s[e] = S_g[(size_t)b * CQK * C_ + e];
  xs_s[tid] = xsum_g[b * C_ + tid];
  if (tid < CQK) ksr_s[tid] = ksum_g[b * CQK + tid];
  __syncthreads();

  const int c = tid;
  const float bvc = bv[c];
  float vs = 0.f;
  float accm[CQK];
#pragma unroll
  for (int m = 0; m < CQK; ++m) accm[m] = 0.f;
  for (int cc = 0; cc < C_; ++cc) {
    const float wvv = wv[(size_t)c * C_ + cc];
    vs += wvv * xs_s[cc];
#pragma unroll
    for (int m = 0; m < CQK; ++m) accm[m] += wvv * S_s[m * C_ + cc];
  }
  vsum_g[b * C_ + c] = vs + (float)NPIX * bvc;
#pragma unroll
  for (int m = 0; m < CQK; ++m)
    kvT_g[((size_t)b * C_ + c) * CQK + m] = (__bf16)(accm[m] + bvc * ksr_s[m]);
  if (tid < CQK) kse_g[b * CQK + tid] = ksr_s[tid] + 1e-6f;
}

// ---------------------------------------------------------------------------
// Pass 3: per 256-pixel tile: out[c,i] = (vsum[c] + kv^T Qn) * 1/(N + Qn.ksum)
// GEMM M=256, N=256, K=32 -> exactly one wmma per 16x16 tile.
// bf16 tiles fetched via ASYNCcnt global->LDS b128 copies.
// ---------------------------------------------------------------------------
__global__ __launch_bounds__(256)
void la_pass3(const __bf16* __restrict__ qn_g, const __bf16* __restrict__ kvT_g,
              const float* __restrict__ vsum_g, const float* __restrict__ kse_g,
              float* __restrict__ out) {
  __shared__ __bf16 qn_s[CQK][NB3 + 8];   // row stride 264 elem = 528 B (16B-mult)
  __shared__ __bf16 kvT_s[C_][40];        // row stride 40 elem  =  80 B (16B-mult)
  __shared__ float  tl_s[NB3];
  __shared__ float  vs_s[C_];
  __shared__ float  ks_s[CQK];
  const int tid  = threadIdx.x;
  const int wave = tid >> 5;
  const int lane = tid & 31;
  const int b    = blockIdx.y;
  const int p0   = blockIdx.x * NB3;

  // async b128 tile fetches: Qn [32 x 256] and kv^T [256 x 32]
  for (int e = tid; e < CQK * (NB3 / 8); e += 256) {   // 1024 chunks
    const int m  = e >> 5;
    const int pc = (e & 31) * 8;
    async_load_b128(&qn_s[m][pc], qn_g + ((size_t)b * CQK + m) * NPIX + p0 + pc);
  }
  for (int e = tid; e < C_ * (CQK / 8); e += 256) {    // 1024 chunks
    const int c  = e >> 2;
    const int mc = (e & 3) * 8;
    async_load_b128(&kvT_s[c][mc], kvT_g + ((size_t)b * C_ + c) * CQK + mc);
  }
  vs_s[tid] = vsum_g[b * C_ + tid];
  if (tid < CQK) ks_s[tid] = kse_g[b * CQK + tid];
  wait_async0();
  __syncthreads();

  {  // taylor denominator per pixel (tid == pixel)
    float s = 0.f;
#pragma unroll
    for (int m = 0; m < CQK; ++m) s += (float)qn_s[m][tid] * ks_s[m];
    tl_s[tid] = 1.f / ((float)NPIX + s);
  }
  __syncthreads();

  for (int t = wave; t < 256; t += 8) {
    const int ctile = t >> 4;
    const int ptile = t & 15;
    v16bf a  = load_A(&kvT_s[ctile * 16][0], 40, lane);          // A: [16 c x 32 m]
    v16bf bm = load_B(&qn_s[0][ptile * 16], NB3 + 8, lane);      // B: [32 m x 16 px]
    v8f acc = {0.f, 0.f, 0.f, 0.f, 0.f, 0.f, 0.f, 0.f};
    acc = wmma_bf16(a, bm, acc);
    const int rowb = (lane >> 4) << 3;
    const int col  = lane & 15;
    const int px   = ptile * 16 + col;
    const float t_ = tl_s[px];
#pragma unroll
    for (int r = 0; r < 8; ++r) {
      const int c = ctile * 16 + rowb + r;
      out[((size_t)b * C_ + c) * NPIX + p0 + px] = (vs_s[c] + acc[r]) * t_;
    }
  }
}

__global__ void la_zero(float* __restrict__ p, int n) {
  const int i = blockIdx.x * blockDim.x + threadIdx.x;
  if (i < n) p[i] = 0.f;
}

// ---------------------------------------------------------------------------
// workspace layout (bytes)
//   0        : S        f32 [B][32][256]   131072
//   131072   : ksum     f32 [B][32]        512
//   131584   : xsum     f32 [B][256]       4096
//   135680   : ksum+eps f32 [B][32]        512
//   136192   : vsum     f32 [B][256]       4096
//   140288   : kv^T     bf16 [B][256][32]  65536
//   205824   : Qn       bf16 [B][32][N]    16777216
// ---------------------------------------------------------------------------
extern "C" void kernel_launch(void* const* d_in, const int* in_sizes, int n_in,
                              void* d_out, int out_size, void* d_ws, size_t ws_size,
                              hipStream_t stream) {
  const float* x  = (const float*)d_in[0];
  const float* wq = (const float*)d_in[1];
  const float* bq = (const float*)d_in[2];
  const float* wk = (const float*)d_in[3];
  const float* bk = (const float*)d_in[4];
  const float* wv = (const float*)d_in[5];
  const float* bv = (const float*)d_in[6];
  float* out = (float*)d_out;

  char* ws = (char*)d_ws;
  float*  S_g    = (float*)(ws + 0);
  float*  ksum_g = (float*)(ws + 131072);
  float*  xsum_g = (float*)(ws + 131584);
  float*  kse_g  = (float*)(ws + 135680);
  float*  vsum_g = (float*)(ws + 136192);
  __bf16* kvT_g  = (__bf16*)(ws + 140288);
  __bf16* qn_g   = (__bf16*)(ws + 205824);

  const int nz = (131072 + 512 + 4096) / 4;  // zero S, ksum, xsum (contiguous)
  la_zero<<<(nz + 255) / 256, 256, 0, stream>>>(S_g, nz);
  la_pass1<<<dim3(NPIX / NB1, B_), 256, 0, stream>>>(x, wq, bq, wk, bk,
                                                     qn_g, S_g, ksum_g, xsum_g);
  la_pass2<<<B_, 256, 0, stream>>>(wv, bv, S_g, ksum_g, xsum_g,
                                   kvT_g, vsum_g, kse_g);
  la_pass3<<<dim3(NPIX / NB3, B_), 256, 0, stream>>>(qn_g, kvT_g, vsum_g, kse_g, out);

  (void)in_sizes; (void)n_in; (void)out_size; (void)ws_size;
}
